// SwinTransformerBlock_13065290514451
// MI455X (gfx1250) — matrix operand order repaired and checked by reading, not compiled, not run on hardware
//
#include <hip/hip_runtime.h>

typedef __attribute__((ext_vector_type(16))) __bf16 v16bf;
typedef __attribute__((ext_vector_type(8)))  float  v8f;

#define DEV __device__ __forceinline__
#define V8ZERO {0.f,0.f,0.f,0.f,0.f,0.f,0.f,0.f}

// ---- problem constants (from reference) ----
constexpr int CC   = 192;          // DIM
constexpr int NH   = 6;            // heads
constexpr int HD   = 32;           // head dim
constexpr int WSZ  = 8;            // window size
constexpr int SS   = 4;            // shift
constexpr int NTOK = WSZ * WSZ;    // 64 tokens / window
constexpr int HP   = 128;          // H = W = 128 (hardcoded, matches harness)
constexpr int NWIN = 8 * (HP/WSZ) * (HP/WSZ);  // 2048 windows total
constexpr int ROWS = NWIN * NTOK;              // 131072 rows
constexpr int HID  = CC * 4;                   // 768
constexpr int QKV_SZ = NWIN * NH * NTOK * HD;  // 25165824 bf16 elems per q/k/v
constexpr float SCALE = 0.17677669529663687f;  // 1/sqrt(32)

// ---- helpers ----
union Frag { unsigned u[8]; uint4 q4[2]; v16bf v; };

DEV unsigned f2bf(float f) {
  unsigned u = __float_as_uint(f);
  return (u + 0x7fffu + ((u >> 16) & 1u)) >> 16;
}
DEV unsigned pk2(float lo, float hi) { return f2bf(lo) | (f2bf(hi) << 16); }

DEV v8f wmma_bf16(const Frag& a, const Frag& b, v8f c) {
  return __builtin_amdgcn_wmma_f32_16x16x32_bf16(
      /*neg_a=*/false, a.v, /*neg_b=*/false, b.v,
      /*c_mod=*/(short)0, c, /*reuse_a=*/false, /*reuse_b=*/false);
}

// A-fragment (16x32 bf16, MxK) from row-major bf16; per-lane data is two
// contiguous 16-byte runs -> two b128 loads. lda even, k0 % 32 == 0,
// base 16B-aligned, row stride multiple of 16B (true for all our buffers).
DEV void load_a(Frag& f, const unsigned short* __restrict__ A,
                int row0, int k0, int lda, int ml, int half) {
  const unsigned short* base = A + (size_t)(row0 + ml) * lda + k0 + (half << 4);
  f.q4[0] = *(const uint4*)(base);
  f.q4[1] = *(const uint4*)(base + 8);
}
// B-fragment (32x16 bf16, KxN) from K-pair-packed dwords [K/2, N].
DEV void load_b(Frag& f, const unsigned* __restrict__ Bp,
                int k0, int n, int N, int half) {
#pragma unroll
  for (int r = 0; r < 8; ++r) {
    int kk = k0 + (half << 4) + (r << 1);
    f.u[r] = Bp[(kk >> 1) * N + n];
  }
}

// ---- weight packing: fp32 [K,N] -> bf16 K-pair-packed dwords [K/2, N] ----
__global__ void pack_w_kernel(const float* __restrict__ w, unsigned* __restrict__ o,
                              int K2, int N) {
  int i = blockIdx.x * blockDim.x + threadIdx.x;
  if (i >= K2 * N) return;
  int k2 = i / N, n = i % N;
  o[i] = pk2(w[(size_t)(2 * k2) * N + n], w[(size_t)(2 * k2 + 1) * N + n]);
}

// ---- LayerNorm (one wave per row) -> bf16 row-major [ROWS, CC] ----
// WINDOWED: output row order = shifted-window-partition order, source = natural x.
template<bool WINDOWED>
__global__ void ln_kernel(const float* __restrict__ x, const float* __restrict__ g,
                          const float* __restrict__ bt, unsigned short* __restrict__ out) {
  int wid = threadIdx.x >> 5, lane = threadIdx.x & 31;
  int row = blockIdx.x * 8 + wid;
  int src;
  if (WINDOWED) {
    int win = row >> 6, t = row & 63;
    int b_ = win >> 8, w_ = win & 255;
    int wi = w_ >> 4, wj = w_ & 15, ti = t >> 3, tj = t & 7;
    int orow = ((wi << 3) + ti + SS) & (HP - 1);
    int ocol = ((wj << 3) + tj + SS) & (HP - 1);
    src = (b_ * HP + orow) * HP + ocol;
  } else {
    src = row;
  }
  const float* xr = x + (size_t)src * CC;
  float v[6]; float s = 0.f, sq = 0.f;
#pragma unroll
  for (int j = 0; j < 6; ++j) { v[j] = xr[lane + 32 * j]; s += v[j]; sq += v[j] * v[j]; }
#pragma unroll
  for (int o = 16; o > 0; o >>= 1) { s += __shfl_xor(s, o, 32); sq += __shfl_xor(sq, o, 32); }
  float mean = s * (1.f / CC);
  float var  = sq * (1.f / CC) - mean * mean;
  float rstd = rsqrtf(var + 1e-5f);
  unsigned short* orp = out + (size_t)row * CC;
#pragma unroll
  for (int j = 0; j < 6; ++j) {
    int c = lane + 32 * j;
    orp[c] = (unsigned short)f2bf((v[j] - mean) * rstd * g[c] + bt[c]);
  }
}

// ---- generic bf16 WMMA GEMM: wave computes a 32x64 tile (8 WMMA / k-step) ----
// MODE 0: out bf16 row-major [M,N] (optional exact GELU)
// MODE 1: QKV scatter -> q/k/v [NWIN, NH, NTOK, HD] bf16 (outb = q base)
// MODE 2: proj: window-reverse + un-roll + residual (yout[idx] = x0[idx] + val)
// MODE 4: fc2: yout[m*CC+n] += val (natural row order)
template<int MODE, bool GELU>
__global__ void gemm_k(const unsigned short* __restrict__ A, const unsigned* __restrict__ Bp,
                       const float* __restrict__ bias, int M, int K, int N,
                       unsigned short* __restrict__ outb, const float* __restrict__ x0,
                       float* __restrict__ yout) {
  int wid = threadIdx.x >> 5, lane = threadIdx.x & 31;
  int half = lane >> 4, ml = lane & 15;
  int gw = blockIdx.x * (blockDim.x >> 5) + wid;
  int ntiles = N >> 6;
  int mtile = gw / ntiles, ntb = gw % ntiles;
  if (mtile >= (M >> 5)) return;
  int row0 = mtile << 5, col0 = ntb << 6;

  v8f zero = V8ZERO;
  v8f acc[2][4];
#pragma unroll
  for (int s2 = 0; s2 < 2; ++s2)
#pragma unroll
    for (int j = 0; j < 4; ++j) acc[s2][j] = zero;

  for (int kb = 0; kb < K; kb += 32) {
    if (kb + 32 < K) {  // prefetch next A k-block (streamed operand)
      __builtin_prefetch(A + (size_t)(row0 + ml) * K + kb + 32);
    }
    Frag a0, a1, b;
    load_a(a0, A, row0,      kb, K, ml, half);
    load_a(a1, A, row0 + 16, kb, K, ml, half);
#pragma unroll
    for (int j = 0; j < 4; ++j) {
      load_b(b, Bp, kb, col0 + (j << 4) + ml, N, half);
      acc[0][j] = wmma_bf16(a0, b, acc[0][j]);
      acc[1][j] = wmma_bf16(a1, b, acc[1][j]);
    }
  }

#pragma unroll
  for (int s2 = 0; s2 < 2; ++s2) {
#pragma unroll
    for (int j = 0; j < 4; ++j) {
#pragma unroll
      for (int i = 0; i < 8; ++i) {
        int m = row0 + (s2 << 4) + i + (half << 3);
        int n = col0 + (j << 4) + ml;
        float val = acc[s2][j][i] + bias[n];
        if constexpr (MODE == 0) {
          if constexpr (GELU) val = 0.5f * val * (1.f + erff(val * 0.70710678118654752f));
          outb[(size_t)m * N + n] = (unsigned short)f2bf(val);
        } else if constexpr (MODE == 1) {
          int which = n / CC, nc = n % CC;
          int h = nc >> 5, d = nc & 31;
          int win = m >> 6, t = m & 63;
          outb[(size_t)which * QKV_SZ + ((size_t)((win * NH + h) * NTOK + t) << 5) + d] =
              (unsigned short)f2bf(val);
        } else if constexpr (MODE == 2) {
          int win = m >> 6, t = m & 63;
          int b_ = win >> 8, w_ = win & 255;
          int wi = w_ >> 4, wj = w_ & 15, ti = t >> 3, tj = t & 7;
          int orow = ((wi << 3) + ti + SS) & (HP - 1);
          int ocol = ((wj << 3) + tj + SS) & (HP - 1);
          size_t idx = (size_t)((b_ * HP + orow) * HP + ocol) * CC + n;
          yout[idx] = x0[idx] + val;
        } else {  // MODE 4
          yout[(size_t)m * CC + n] += val;
        }
      }
    }
  }
}

// ---- attention: one wave per (window, head, 16-query tile) ----
__global__ void attn_k(const unsigned short* __restrict__ qkv,
                       const float* __restrict__ btab,
                       unsigned short* __restrict__ aout) {
  __shared__ __align__(16) unsigned short probs[8][16][64];
  int wid = threadIdx.x >> 5, lane = threadIdx.x & 31;
  int half = lane >> 4, ml = lane & 15;
  int gw = blockIdx.x * 8 + wid;
  int unit = gw >> 2, mtile = gw & 3;
  int win = unit / NH, h = unit % NH;
  const unsigned short* qb  = qkv + (size_t)(win * NH + h) * NTOK * HD;
  const unsigned short* kbp = qb + (size_t)QKV_SZ;
  const unsigned short* vbp = qb + (size_t)2 * QKV_SZ;

  // scores: q (16x32) @ k^T (32x64) -> 4 WMMA tiles
  Frag aq; load_a(aq, qb, mtile << 4, 0, HD, ml, half);
  v8f s[4];
#pragma unroll
  for (int j = 0; j < 4; ++j) {
    Frag bk;
    // per-lane B data = 16 contiguous head-dim bf16 of key token n -> 2 x b128
    const unsigned short* kb_base = kbp + (size_t)((j << 4) + ml) * HD + (half << 4);
    bk.q4[0] = *(const uint4*)(kb_base);
    bk.q4[1] = *(const uint4*)(kb_base + 8);
    v8f z = V8ZERO;
    s[j] = wmma_bf16(aq, bk, z);
  }

  // scale + relative-position bias + shift mask
  int w_ = win & 255;
  int wi = w_ >> 4, wj = w_ & 15;
  float p[4][8];
#pragma unroll
  for (int j = 0; j < 4; ++j) {
#pragma unroll
    for (int i = 0; i < 8; ++i) {
      int mq = (mtile << 4) + i + (half << 3);  // query token 0..63
      int nk = (j << 4) + ml;                   // key token 0..63
      int qi = mq >> 3, qj = mq & 7, ki = nk >> 3, kj = nk & 7;
      int rel = (qi - ki + 7) * 15 + (qj - kj + 7);
      float bv = btab[rel * NH + h];
      int oqr = ((wi << 3) + qi + SS) & (HP - 1), oqc = ((wj << 3) + qj + SS) & (HP - 1);
      int okr = ((wi << 3) + ki + SS) & (HP - 1), okc = ((wj << 3) + kj + SS) & (HP - 1);
      int cq = (oqr < 120 ? 0 : (oqr < 124 ? 1 : 2)) * 3 + (oqc < 120 ? 0 : (oqc < 124 ? 1 : 2));
      int ck = (okr < 120 ? 0 : (okr < 124 ? 1 : 2)) * 3 + (okc < 120 ? 0 : (okc < 124 ? 1 : 2));
      p[j][i] = s[j][i] * SCALE + bv + (cq == ck ? 0.f : -100.f);
    }
  }

  // softmax over 64 keys per row (row lives in one 16-lane half across 4 tiles)
#pragma unroll
  for (int i = 0; i < 8; ++i) {
    float mx = fmaxf(fmaxf(p[0][i], p[1][i]), fmaxf(p[2][i], p[3][i]));
#pragma unroll
    for (int o = 8; o > 0; o >>= 1) mx = fmaxf(mx, __shfl_xor(mx, o, 16));
    float sm = 0.f;
#pragma unroll
    for (int j = 0; j < 4; ++j) { p[j][i] = __expf(p[j][i] - mx); sm += p[j][i]; }
#pragma unroll
    for (int o = 8; o > 0; o >>= 1) sm += __shfl_xor(sm, o, 16);
    float inv = 1.f / sm;
#pragma unroll
    for (int j = 0; j < 4; ++j)
      probs[wid][i + (half << 3)][(j << 4) + ml] = (unsigned short)f2bf(p[j][i] * inv);
  }
  __syncthreads();

  // out = probs (16x64) @ v (64x32): 2 N-tiles x 2 K-steps
#pragma unroll
  for (int t2 = 0; t2 < 2; ++t2) {
    v8f o8 = V8ZERO;
#pragma unroll
    for (int ks = 0; ks < 2; ++ks) {
      Frag ap;
      // per-lane A data = two contiguous 16B runs of this lane's probs row
      const unsigned short* pr = &probs[wid][ml][0] + (ks << 5) + (half << 4);
      ap.q4[0] = *(const uint4*)(pr);
      ap.q4[1] = *(const uint4*)(pr + 8);
      Frag bv;
#pragma unroll
      for (int r = 0; r < 8; ++r) {
        int kt = (ks << 5) + (half << 4) + (r << 1);  // key token (pair along K)
        int nd = (t2 << 4) + ml;                      // head-dim column
        unsigned lo = vbp[(size_t)kt * HD + nd];
        unsigned hi = vbp[(size_t)(kt + 1) * HD + nd];
        bv.u[r] = lo | (hi << 16);
      }
      o8 = wmma_bf16(ap, bv, o8);
    }
#pragma unroll
    for (int i = 0; i < 8; ++i) {
      int mq = (mtile << 4) + i + (half << 3);
      int nd = (t2 << 4) + ml;
      size_t grow = (size_t)win * NTOK + mq;
      aout[grow * CC + h * HD + nd] = (unsigned short)f2bf(o8[i]);
    }
  }
}

// ---- host-side orchestration ----
extern "C" void kernel_launch(void* const* d_in, const int* in_sizes, int n_in,
                              void* d_out, int out_size, void* d_ws, size_t ws_size,
                              hipStream_t stream) {
  (void)in_sizes; (void)n_in; (void)out_size; (void)ws_size;
  const float* x     = (const float*)d_in[0];
  const float* n1g   = (const float*)d_in[3];
  const float* n1b   = (const float*)d_in[4];
  const float* qkvw  = (const float*)d_in[5];
  const float* qkvb  = (const float*)d_in[6];
  const float* projw = (const float*)d_in[7];
  const float* projb = (const float*)d_in[8];
  const float* relt  = (const float*)d_in[9];
  const float* n2g   = (const float*)d_in[10];
  const float* n2b   = (const float*)d_in[11];
  const float* fc1w  = (const float*)d_in[12];
  const float* fc1b  = (const float*)d_in[13];
  const float* fc2w  = (const float*)d_in[14];
  const float* fc2b  = (const float*)d_in[15];
  float* out = (float*)d_out;

  char* ws = (char*)d_ws;
  // packed weights
  unsigned* qkvWp = (unsigned*)(ws + 0);         // 96*576 dwords
  unsigned* projWp = (unsigned*)(ws + 221184);   // 96*192 dwords
  unsigned* fc1Wp  = (unsigned*)(ws + 294912);   // 96*768 dwords
  unsigned* fc2Wp  = (unsigned*)(ws + 589824);   // 384*192 dwords
  // activation buffers (bf16)
  const size_t OFF_XW   = 1048576;                             // 50.3 MB: LN1 windowed rows
  const size_t OFF_QKV  = OFF_XW + (size_t)ROWS * CC * 2;      // 151 MB: q/k/v
  const size_t OFF_ATTN = OFF_QKV + (size_t)3 * QKV_SZ * 2;    // 50.3 MB: attn out
  unsigned short* xw    = (unsigned short*)(ws + OFF_XW);
  unsigned short* qkvB  = (unsigned short*)(ws + OFF_QKV);
  unsigned short* attnB = (unsigned short*)(ws + OFF_ATTN);
  unsigned short* ln2B  = xw;                       // reuse (xw dead after QKV GEMM)
  unsigned short* hB    = qkvB;                     // reuse qkv+attn regions (201 MB)

  // 1) pack weights to bf16 K-pair layout
  pack_w_kernel<<<(96 * 576 + 255) / 256, 256, 0, stream>>>(qkvw, qkvWp, 96, 576);
  pack_w_kernel<<<(96 * 192 + 255) / 256, 256, 0, stream>>>(projw, projWp, 96, 192);
  pack_w_kernel<<<(96 * 768 + 255) / 256, 256, 0, stream>>>(fc1w, fc1Wp, 96, 768);
  pack_w_kernel<<<(384 * 192 + 255) / 256, 256, 0, stream>>>(fc2w, fc2Wp, 384, 192);

  // 2) LN1 + shift + window partition
  ln_kernel<true><<<ROWS / 8, 256, 0, stream>>>(x, n1g, n1b, xw);

  // 3) QKV GEMM (131072 x 192 x 576), scatter per (window, head)
  gemm_k<1, false><<<(ROWS / 32) * (576 / 64) / 8, 256, 0, stream>>>(
      xw, qkvWp, qkvb, ROWS, CC, 3 * CC, qkvB, nullptr, nullptr);

  // 4) windowed attention (12288 units x 4 query-tiles = 49152 waves)
  attn_k<<<(NWIN * NH * 4) / 8, 256, 0, stream>>>(qkvB, relt, attnB);

  // 5) proj GEMM + window-reverse + un-roll + residual -> d_out (fp32)
  gemm_k<2, false><<<(ROWS / 32) * (CC / 64) / 8, 256, 0, stream>>>(
      attnB, projWp, projb, ROWS, CC, CC, nullptr, x, out);

  // 6) LN2 (natural order) -> bf16
  ln_kernel<false><<<ROWS / 8, 256, 0, stream>>>(out, n2g, n2b, ln2B);

  // 7) fc1 GEMM + exact GELU -> bf16
  gemm_k<0, true><<<(ROWS / 32) * (HID / 64) / 8, 256, 0, stream>>>(
      ln2B, fc1Wp, fc1b, ROWS, CC, HID, hB, nullptr, nullptr);

  // 8) fc2 GEMM, accumulate into d_out
  gemm_k<4, false><<<(ROWS / 32) * (CC / 64) / 8, 256, 0, stream>>>(
      hB, fc2Wp, fc2b, ROWS, HID, CC, nullptr, nullptr, out);
}
